// Head_5128190951491
// MI455X (gfx1250) — matrix operand order repaired and checked by reading, not compiled, not run on hardware
//
#include <hip/hip_runtime.h>
#include <hip/hip_bf16.h>
#include <math.h>

typedef __attribute__((ext_vector_type(2)))  float  v2f;
typedef __attribute__((ext_vector_type(8)))  float  v8f;
typedef __attribute__((ext_vector_type(8)))  __bf16 v8bf;
typedef __attribute__((ext_vector_type(16))) __bf16 v16bf;

#define NBATCH 4
#define T_SEQ  4096
#define C_DIM  512
#define D_DIM  64

#define XS_STRIDE 516   // fp32 x-tile rows: 16B-aligned, conflict-free b64 reads
#define KS_STRIDE 72    // bf16 K/V chunk rows: 144B = 16B-aligned, banks 36n%64 distinct
#define PS_STRIDE 40    // bf16 P stash rows: 80B = 16B-aligned, banks 20m%64 distinct

__device__ __forceinline__ float arnold_map(float t, float kcoef) {
    // (t + OMEGA - |K|/(2pi) * sin(2pi t)) mod 1 ; jnp.mod -> u - floor(u)
    float u = t + 0.618f - kcoef * __sinf(6.28318530717958647692f * t);
    return u - floorf(u);
}

__device__ __forceinline__ v16bf ld16(const __bf16* p0, const __bf16* p1) {
    v8bf a = *(const v8bf*)p0;   // 16B
    v8bf b = *(const v8bf*)p1;   // 16B
    return __builtin_shufflevector(a, b, 0,1,2,3,4,5,6,7,8,9,10,11,12,13,14,15);
}

// ---------------------------------------------------------------------------
// Kernel 1: q/k/v projections (fp32 WMMA) + Arnold map; emit bf16 q,k,v.
// One block = one 16-row tile of flattened [B*T]; wave w does d-column tile w.
// ---------------------------------------------------------------------------
__global__ __launch_bounds__(128) void proj_kernel(
    const float* __restrict__ x,  const float* __restrict__ Wq,
    const float* __restrict__ Wk, const float* __restrict__ Wv,
    const float* __restrict__ Kparam,
    __bf16* __restrict__ qo, __bf16* __restrict__ ko, __bf16* __restrict__ vo)
{
    __shared__ float xs[16 * XS_STRIDE];

    const int tid  = threadIdx.x;
    const int row0 = blockIdx.x * 16;

    for (int j = tid; j < 16 * 128; j += 128) {
        const int r  = j >> 7;
        const int c4 = (j & 127) * 4;
        const float4 d = *(const float4*)(x + (size_t)(row0 + r) * C_DIM + c4);
        float* dst = &xs[r * XS_STRIDE + c4];
        dst[0] = d.x; dst[1] = d.y; dst[2] = d.z; dst[3] = d.w;
    }
    __syncthreads();

    const int wave = tid >> 5;
    const int lane = tid & 31;
    const int m    = lane & 15;
    const int kb   = (lane >> 4) * 2;
    const int d0   = wave * 16;

    v8f accq = {}, acck = {}, accv = {};

    const float* wqp = Wq + (size_t)(d0 + m) * C_DIM + kb;
    const float* wkp = Wk + (size_t)(d0 + m) * C_DIM + kb;
    const float* wvp = Wv + (size_t)(d0 + m) * C_DIM + kb;
    const float* xp  = &xs[m * XS_STRIDE + kb];

    for (int c = 0; c < C_DIM; c += 4) {
        const v2f xa = *(const v2f*)(xp + c);
        const v2f bq = *(const v2f*)(wqp + c);
        const v2f bk = *(const v2f*)(wkp + c);
        const v2f bv = *(const v2f*)(wvp + c);
        accq = __builtin_amdgcn_wmma_f32_16x16x4_f32(false, xa, false, bq, (short)0, accq, false, false);
        acck = __builtin_amdgcn_wmma_f32_16x16x4_f32(false, xa, false, bk, (short)0, acck, false, false);
        accv = __builtin_amdgcn_wmma_f32_16x16x4_f32(false, xa, false, bv, (short)0, accv, false, false);
    }

    const float kcoef = fabsf(Kparam[0]) * 0.15915494309189533577f; // |K|/(2pi)
    const int hi = lane >> 4;
    #pragma unroll
    for (int r = 0; r < 8; ++r) {
        const size_t orow = (size_t)(row0 + r + 8 * hi);
        qo[orow * D_DIM + d0 + m] = (__bf16)arnold_map(accq[r], kcoef);
        ko[orow * D_DIM + d0 + m] = (__bf16)arnold_map(acck[r], kcoef);
        vo[orow * D_DIM + d0 + m] = (__bf16)accv[r];
    }
}

// ---------------------------------------------------------------------------
// Kernel 2: bf16 flash attention, fp32 accum/softmax. 8 waves/block, one
// 16-row query tile per wave. 64-key chunks staged in LDS (K natural,
// V transposed); 32 keys per online-softmax pass; 8 bf16 WMMA per pass.
// ---------------------------------------------------------------------------
__global__ __launch_bounds__(256) void attn_kernel(
    const __bf16* __restrict__ q, const __bf16* __restrict__ k,
    const __bf16* __restrict__ v, float* __restrict__ out)
{
    __shared__ __bf16 ks [64 * KS_STRIDE];       // K chunk  [key][d]
    __shared__ __bf16 vts[64 * KS_STRIDE];       // V^T chunk [d][key]
    __shared__ __bf16 ps [8 * 16 * PS_STRIDE];   // per-wave bf16 P stash

    const int tid  = threadIdx.x;
    const int wave = tid >> 5;
    const int lane = tid & 31;
    const int m    = lane & 15;
    const int hi   = lane >> 4;

    const int blocks_per_batch = T_SEQ / 128;              // 32
    const int b  = blockIdx.x / blocks_per_batch;
    const int q0 = (blockIdx.x % blocks_per_batch) * 128 + wave * 16;

    // q tile in bf16 A-layout: per lane row m, K contiguous at 16*hi
    const __bf16* qbase = q + ((size_t)b * T_SEQ + q0 + m) * D_DIM;
    const v16bf qa0 = ld16(qbase +      16 * hi, qbase +  8 + 16 * hi);  // d 0..31
    const v16bf qa1 = ld16(qbase + 32 + 16 * hi, qbase + 40 + 16 * hi);  // d 32..63

    v8f acc0 = {}, acc1 = {}, acc2 = {}, acc3 = {};
    float mrow[8], lrow[8];
    #pragma unroll
    for (int r = 0; r < 8; ++r) { mrow[r] = -INFINITY; lrow[r] = 0.0f; }

    __bf16* psw = &ps[wave * 16 * PS_STRIDE];

    for (int k0 = 0; k0 < T_SEQ; k0 += 64) {
        __syncthreads();
        const __bf16* kg = k + ((size_t)b * T_SEQ + k0) * D_DIM;
        const __bf16* vg = v + ((size_t)b * T_SEQ + k0) * D_DIM;
        // stage: K natural copy (16B chunks), V transposed (b16 scatter)
        for (int j = tid; j < 64 * 8; j += 256) {
            const int key = j >> 3;
            const int c8  = (j & 7) * 8;
            *(v8bf*)(&ks[key * KS_STRIDE + c8]) = *(const v8bf*)(kg + key * D_DIM + c8);
            const v8bf vv = *(const v8bf*)(vg + key * D_DIM + c8);
            #pragma unroll
            for (int i = 0; i < 8; ++i) vts[(c8 + i) * KS_STRIDE + key] = vv[i];
        }
        __syncthreads();

        #pragma unroll
        for (int ktt = 0; ktt < 2; ++ktt) {
            const int kB = ktt * 32;                 // key offset within chunk

            // ---- S[16x32] = Q @ K^T : 4 bf16 WMMA --------------------------
            v8f S0 = {}, S1 = {};
            {
                const __bf16* kr0 = &ks[(kB +      m) * KS_STRIDE + 16 * hi];
                const __bf16* kr1 = &ks[(kB + 16 + m) * KS_STRIDE + 16 * hi];
                const v16bf b00 = ld16(kr0,      kr0 + 8);        // keys n,    d 0..31
                const v16bf b01 = ld16(kr0 + 32, kr0 + 40);       // keys n,    d 32..63
                const v16bf b10 = ld16(kr1,      kr1 + 8);        // keys n+16, d 0..31
                const v16bf b11 = ld16(kr1 + 32, kr1 + 40);       // keys n+16, d 32..63
                S0 = __builtin_amdgcn_wmma_f32_16x16x32_bf16(false, qa0, false, b00, (short)0, S0, false, false);
                S0 = __builtin_amdgcn_wmma_f32_16x16x32_bf16(false, qa1, false, b01, (short)0, S0, false, false);
                S1 = __builtin_amdgcn_wmma_f32_16x16x32_bf16(false, qa0, false, b10, (short)0, S1, false, false);
                S1 = __builtin_amdgcn_wmma_f32_16x16x32_bf16(false, qa1, false, b11, (short)0, S1, false, false);
            }

            // ---- online softmax over 32 keys (fp32) ------------------------
            float cfv[8];
            #pragma unroll
            for (int r = 0; r < 8; ++r) {
                const float s0 = S0[r] * 0.125f;     // 1/sqrt(64)
                const float s1 = S1[r] * 0.125f;
                float tmax = fmaxf(s0, s1);
                tmax = fmaxf(tmax, __shfl_xor(tmax, 1, 32));
                tmax = fmaxf(tmax, __shfl_xor(tmax, 2, 32));
                tmax = fmaxf(tmax, __shfl_xor(tmax, 4, 32));
                tmax = fmaxf(tmax, __shfl_xor(tmax, 8, 32));
                const float mnew = fmaxf(mrow[r], tmax);
                const float cf   = __expf(mrow[r] - mnew);
                const float p0   = __expf(s0 - mnew);
                const float p1   = __expf(s1 - mnew);
                mrow[r] = mnew;
                float rs = p0 + p1;
                rs += __shfl_xor(rs, 1, 32);
                rs += __shfl_xor(rs, 2, 32);
                rs += __shfl_xor(rs, 4, 32);
                rs += __shfl_xor(rs, 8, 32);
                lrow[r] = lrow[r] * cf + rs;
                cfv[r]  = cf;
                // stash P as bf16 (C layout write)
                psw[(r + 8 * hi) * PS_STRIDE +      m] = (__bf16)p0;
                psw[(r + 8 * hi) * PS_STRIDE + 16 + m] = (__bf16)p1;
            }
            #pragma unroll
            for (int r = 0; r < 8; ++r) {
                acc0[r] *= cfv[r]; acc1[r] *= cfv[r];
                acc2[r] *= cfv[r]; acc3[r] *= cfv[r];
            }

            // ---- acc += P @ V : 4 bf16 WMMA --------------------------------
            const __bf16* prow = &psw[m * PS_STRIDE];
            const v16bf pa = ld16(prow + 16 * hi, prow + 8 + 16 * hi);   // P[m, key 0..31]
            const __bf16* vr0 = &vts[( 0 + m) * KS_STRIDE + kB + 16 * hi];
            const __bf16* vr1 = &vts[(16 + m) * KS_STRIDE + kB + 16 * hi];
            const __bf16* vr2 = &vts[(32 + m) * KS_STRIDE + kB + 16 * hi];
            const __bf16* vr3 = &vts[(48 + m) * KS_STRIDE + kB + 16 * hi];
            const v16bf bv0 = ld16(vr0, vr0 + 8);
            const v16bf bv1 = ld16(vr1, vr1 + 8);
            const v16bf bv2 = ld16(vr2, vr2 + 8);
            const v16bf bv3 = ld16(vr3, vr3 + 8);
            acc0 = __builtin_amdgcn_wmma_f32_16x16x32_bf16(false, pa, false, bv0, (short)0, acc0, false, false);
            acc1 = __builtin_amdgcn_wmma_f32_16x16x32_bf16(false, pa, false, bv1, (short)0, acc1, false, false);
            acc2 = __builtin_amdgcn_wmma_f32_16x16x32_bf16(false, pa, false, bv2, (short)0, acc2, false, false);
            acc3 = __builtin_amdgcn_wmma_f32_16x16x32_bf16(false, pa, false, bv3, (short)0, acc3, false, false);
        }
    }

    // normalize and store fp32 output
    float* obase = out + ((size_t)b * T_SEQ + q0) * D_DIM;
    #pragma unroll
    for (int r = 0; r < 8; ++r) {
        const float inv = 1.0f / lrow[r];
        const size_t orow = (size_t)(r + 8 * hi) * D_DIM;
        obase[orow +  0 + m] = acc0[r] * inv;
        obase[orow + 16 + m] = acc1[r] * inv;
        obase[orow + 32 + m] = acc2[r] * inv;
        obase[orow + 48 + m] = acc3[r] * inv;
    }
}

// ---------------------------------------------------------------------------
extern "C" void kernel_launch(void* const* d_in, const int* in_sizes, int n_in,
                              void* d_out, int out_size, void* d_ws, size_t ws_size,
                              hipStream_t stream) {
    (void)in_sizes; (void)n_in; (void)out_size; (void)ws_size;
    const float* x  = (const float*)d_in[0];
    const float* Wq = (const float*)d_in[1];
    const float* Wk = (const float*)d_in[2];
    const float* Wv = (const float*)d_in[3];
    const float* Kp = (const float*)d_in[4];
    float* out = (float*)d_out;

    const size_t nqkv = (size_t)NBATCH * T_SEQ * D_DIM;   // 1,048,576 elems each
    __bf16* qws = (__bf16*)d_ws;
    __bf16* kws = qws + nqkv;
    __bf16* vws = kws + nqkv;

    proj_kernel<<<(NBATCH * T_SEQ) / 16, 128, 0, stream>>>(x, Wq, Wk, Wv, Kp, qws, kws, vws);
    attn_kernel<<<(NBATCH * T_SEQ) / 128, 256, 0, stream>>>(qws, kws, vws, out);
}
